// LiftedStructureLoss_403726926379
// MI455X (gfx1250) — compile-verified, matrix operand order checked by reading, not run on hardware
//
#include <hip/hip_runtime.h>
#include <hip/hip_bf16.h>
#include <stdint.h>

// Problem constants (from reference)
#define N_TOT   8192
#define D_DIM   512
#define BM      128      // rows per workgroup (8 waves x 16-row strips)
#define TILE    16
#define CBLK    128      // columns staged per TDM block
#define ROWPITCH_B 1040  // 512 bf16 (1024B) + 16B TDM pad -> 260 dwords, conflict-free
#define BUFBYTES (CBLK * ROWPITCH_B)   // 133120 B per buffer

typedef __attribute__((ext_vector_type(16))) __bf16 bf16x16;
typedef __attribute__((ext_vector_type(8)))  __bf16 bf16x8;
typedef __attribute__((ext_vector_type(8)))  float  f32x8;
typedef __attribute__((ext_vector_type(4)))  unsigned int uint32x4;
typedef __attribute__((ext_vector_type(8)))  int int32x8;
typedef __attribute__((ext_vector_type(4)))  int int32x4;

union AFrag { bf16x16 v; bf16x8 h[2]; };

// ---------------------------------------------------------------------------
// Kernel 1: fp32 -> bf16 (round-to-nearest-even), 4 elems/thread
// ---------------------------------------------------------------------------
__global__ __launch_bounds__(256) void cvt_bf16_kernel(
    const float* __restrict__ x, uint16_t* __restrict__ xb)
{
    int i = (blockIdx.x * 256 + threadIdx.x) * 4;   // grid sized so i+3 < N*D
    float4 f = *(const float4*)(x + i);
    uint16_t r[4];
    const float fv[4] = {f.x, f.y, f.z, f.w};
    #pragma unroll
    for (int j = 0; j < 4; ++j) {
        uint32_t b = __float_as_uint(fv[j]);
        b += 0x7FFFu + ((b >> 16) & 1u);            // RNE
        r[j] = (uint16_t)(b >> 16);
    }
    *(ushort4*)(xb + i) = make_ushort4(r[0], r[1], r[2], r[3]);
}

// ---------------------------------------------------------------------------
// TDM: DMA one 128-row x 512-K bf16 tile of X into LDS, padded 16B per row.
// D# per CDNA5 ISA ch.8: group0 {count, lds_addr, global_addr, type=2},
// group1 {data_size=2B, pad_enable, pad_interval=7(256dw), pad_amount=3(4dw),
//         tensor_dim0=512, tensor_dim1=8192, tile_dim0=512, tile_dim1=128,
//         tensor_dim0_stride=512}, groups 2/3 zero (2-D tile).
// This toolchain uses the 6-arg builtin form (extra int32x8 before cpol).
// ---------------------------------------------------------------------------
__device__ __forceinline__ void tdm_load_tile(uint32_t lds_off, uint64_t gaddr)
{
    uint32x4 g0 = {
        1u,                                                     // count=1 (valid user D#)
        lds_off,                                                // lds_addr (bytes)
        (uint32_t)gaddr,                                        // global_addr[31:0]
        ((uint32_t)(gaddr >> 32) & 0x01FFFFFFu) | (2u << 30)    // global_addr[56:32] | type=2
    };
    int32x8 g1 = {
        (int)((1u << 16) | (1u << 20) | (7u << 22) | (3u << 25)), // data_size=1(2B), pad_en, intv=7, amt=3
        (int)(512u << 16),     // [63:48] tensor_dim0 lo16 = 512
        (int)(8192u << 16),    // [95:80] tensor_dim1 lo16 = 8192
        (int)(512u << 16),     // [127:112] tile_dim0 = 512
        128,                   // [143:128] tile_dim1 = 128 ; tile_dim2 = 0
        512,                   // [191:160] tensor_dim0_stride lo32 = 512
        0, 0                   // stride hi / tensor_dim1_stride = 0 (unused, 2-D tile)
    };
    int32x4 gz4 = {0, 0, 0, 0};
    int32x8 gz8 = {0, 0, 0, 0, 0, 0, 0, 0};
    __builtin_amdgcn_tensor_load_to_lds(g0, g1, gz4, gz4, gz8, 0);
}

// ---------------------------------------------------------------------------
// Kernel 2: fused Gram-matrix + masked exp row sums.
// Workgroup = 8 waves x 16-row strips (128 rows). B blocks of 128 columns are
// TDM-staged into double-buffered LDS and shared by all 8 waves; A strips are
// register-hoisted by the compiler. WMMA bf16 16x16x32, f32 accumulate.
// ---------------------------------------------------------------------------
__global__ __launch_bounds__(256) void gram_loss_kernel(
    const uint16_t* __restrict__ xb_raw, const int* __restrict__ tgt,
    float* __restrict__ pos_sum, float* __restrict__ neg_sum,
    float* __restrict__ last_stats)
{
    extern __shared__ char smem[];                      // 2 * BUFBYTES
    const __bf16* xb = (const __bf16*)xb_raw;
    const int lane = threadIdx.x & 31;
    const int wave = threadIdx.x >> 5;
    const int l16  = lane & 15;
    const int hi   = lane >> 4;                         // 0/1: K-half & row-half select
    const int rowStrip = blockIdx.x * BM + wave * TILE;

    const uint32_t ldsBase = (uint32_t)(uintptr_t)(&smem[0]);   // LDS byte offset
    const uint64_t gbase   = (uint64_t)(uintptr_t)xb_raw;

    // A-fragment row pointer (16-bit A layout: lane%16 = M; lane/16 picks K-halves)
    const __bf16* aRow = xb + (size_t)(rowStrip + l16) * D_DIM;

    int trow[8];
    #pragma unroll
    for (int j = 0; j < 8; ++j)
        trow[j] = tgt[rowStrip + hi * 8 + j];

    float posAcc[8], negAcc[8];
    #pragma unroll
    for (int j = 0; j < 8; ++j) { posAcc[j] = 0.0f; negAcc[j] = 0.0f; }

    const bool lastWave = (rowStrip + TILE == N_TOT);   // strip containing row N-1
    float lpSim = 0.0f, lpCnt = 0.0f, lnSim = 0.0f, lnCnt = 0.0f;

    const int NBLK = N_TOT / CBLK;                      // 64 column blocks

    if (wave == 0)                                      // prologue: stage block 0
        tdm_load_tile(ldsBase, gbase);

    for (int cb = 0; cb < NBLK; ++cb) {
        if (wave == 0) {
            if (cb + 1 < NBLK) {                        // stage next block, overlap w/ compute
                tdm_load_tile(ldsBase + (uint32_t)((cb + 1) & 1) * BUFBYTES,
                              gbase + (uint64_t)(cb + 1) * CBLK * (D_DIM * 2));
                __builtin_amdgcn_s_wait_tensorcnt(1);   // in-order TDM => block cb landed
            } else {
                __builtin_amdgcn_s_wait_tensorcnt(0);
            }
        }
        __syncthreads();                                // publish LDS tile to all waves

        const char* bBlock = smem + (size_t)(cb & 1) * BUFBYTES;

        for (int sub = 0; sub < CBLK / TILE; ++sub) {
            const int colBase = cb * CBLK + sub * TILE;
            const __bf16* bCol = (const __bf16*)(bBlock + (size_t)(sub * TILE + l16) * ROWPITCH_B);
            const int tcol = tgt[colBase + l16];
            const int col  = colBase + l16;

            f32x8 c;
            #pragma unroll
            for (int j = 0; j < 8; ++j) c[j] = 0.0f;

            #pragma unroll
            for (int k = 0; k < D_DIM; k += 32) {
                AFrag a, b;
                // lanes 0-15: K=k+0..7 & k+16..23 ; lanes 16-31: K=k+8..15 & k+24..31
                a.h[0] = *(const bf16x8*)(aRow + k + hi * 8);
                a.h[1] = *(const bf16x8*)(aRow + k + hi * 8 + 16);
                b.h[0] = *(const bf16x8*)(bCol + k + hi * 8);        // ds_load_b128
                b.h[1] = *(const bf16x8*)(bCol + k + hi * 8 + 16);
                c = __builtin_amdgcn_wmma_f32_16x16x32_bf16(
                        false, a.v, false, b.v, (short)0, c, false, false);
            }

            #pragma unroll
            for (int j = 0; j < 8; ++j) {
                const int row = rowStrip + hi * 8 + j;
                const float s = c[j];
                const bool same   = (trow[j] == tcol);
                const bool isSelf = (row == col);
                const bool pos = same && !isSelf && (s < 1.0f);
                const bool neg = !same;
                const float ep = __expf(1.0f - s);
                const float en = __expf(s);
                posAcc[j] += pos ? ep : 0.0f;
                negAcc[j] += neg ? en : 0.0f;
                if (lastWave && hi == 1 && j == 7) {    // row == N_TOT-1
                    lpSim += pos ? s : 0.0f;  lpCnt += pos ? 1.0f : 0.0f;
                    lnSim += neg ? s : 0.0f;  lnCnt += neg ? 1.0f : 0.0f;
                }
            }
        }
        __syncthreads();                                // all reads done before buffer reuse
    }

    // Reduce across the 16 lanes sharing each row (groups {0..15}, {16..31})
    #pragma unroll
    for (int j = 0; j < 8; ++j) {
        #pragma unroll
        for (int m = 8; m >= 1; m >>= 1) {
            posAcc[j] += __shfl_xor(posAcc[j], m, 16);
            negAcc[j] += __shfl_xor(negAcc[j], m, 16);
        }
    }
    if (l16 == 0) {
        #pragma unroll
        for (int j = 0; j < 8; ++j) {
            const int row = rowStrip + hi * 8 + j;
            pos_sum[row] = posAcc[j];
            neg_sum[row] = negAcc[j];
        }
    }

    if (lastWave && hi == 1) {                          // lanes 16..31 all active
        #pragma unroll
        for (int m = 8; m >= 1; m >>= 1) {
            lpSim += __shfl_xor(lpSim, m, 16);
            lpCnt += __shfl_xor(lpCnt, m, 16);
            lnSim += __shfl_xor(lnSim, m, 16);
            lnCnt += __shfl_xor(lnCnt, m, 16);
        }
        if (l16 == 0) {
            last_stats[0] = lpSim; last_stats[1] = lpCnt;
            last_stats[2] = lnSim; last_stats[3] = lnCnt;
        }
    }
}

// ---------------------------------------------------------------------------
// Kernel 3: deterministic final reduction -> 4 scalars
// ---------------------------------------------------------------------------
__global__ __launch_bounds__(256) void finalize_kernel(
    const float* __restrict__ pos_sum, const float* __restrict__ neg_sum,
    const float* __restrict__ last_stats, float* __restrict__ out)
{
    __shared__ float sLoss[256];
    __shared__ float sSkip[256];
    const int tid = threadIdx.x;
    float loss = 0.0f, skip = 0.0f;
    for (int r = tid; r < N_TOT; r += 256) {
        const float ps = pos_sum[r];
        const float ns = neg_sum[r];
        if (ns > 0.0f) loss += __logf(ps) + __logf(ns);   // rows w/o negatives skipped
        else           skip += 1.0f;
    }
    sLoss[tid] = loss; sSkip[tid] = skip;
    __syncthreads();
    for (int s = 128; s > 0; s >>= 1) {
        if (tid < s) { sLoss[tid] += sLoss[tid + s]; sSkip[tid] += sSkip[tid + s]; }
        __syncthreads();
    }
    if (tid == 0) {
        out[0] = sLoss[0] / (float)N_TOT;               // loss
        out[1] = sSkip[0] / (float)N_TOT;               // prec (fraction skipped)
        out[2] = last_stats[0] / last_stats[1];         // mean_pos_sim (last row)
        out[3] = last_stats[2] / last_stats[3];         // mean_neg_sim (last row)
    }
}

// ---------------------------------------------------------------------------
extern "C" void kernel_launch(void* const* d_in, const int* in_sizes, int n_in,
                              void* d_out, int out_size, void* d_ws, size_t ws_size,
                              hipStream_t stream) {
    const float* x   = (const float*)d_in[0];
    const int*   tgt = (const int*)d_in[1];
    // d_in[2] = nonorm scalar, unused (default path)
    float* out = (float*)d_out;

    // Workspace: [ bf16 X : N*D*2 B ][ pos_sum : N f32 ][ neg_sum : N f32 ][ last_stats : 4 f32 ]
    uint16_t* xb      = (uint16_t*)d_ws;
    float* pos_sum    = (float*)((char*)d_ws + (size_t)N_TOT * D_DIM * sizeof(uint16_t));
    float* neg_sum    = pos_sum + N_TOT;
    float* last_stats = neg_sum + N_TOT;

    cvt_bf16_kernel<<<(N_TOT * D_DIM) / (256 * 4), 256, 0, stream>>>(x, xb);

    // 64 workgroups x 8 waves; 2 x 130KB LDS double buffer (<= 320KB/WGP)
    gram_loss_kernel<<<N_TOT / BM, 256, 2 * BUFBYTES, stream>>>(
        xb, tgt, pos_sum, neg_sum, last_stats);

    finalize_kernel<<<1, 256, 0, stream>>>(pos_sum, neg_sum, last_stats, out);
}
